// Encoder_86835648791161
// MI455X (gfx1250) — matrix operand order repaired and checked by reading, not compiled, not run on hardware
//
#include <hip/hip_runtime.h>
#include <cstdint>
#include <cstddef>

// ---------------- problem constants ----------------
constexpr int BATCH = 4;
constexpr int SEQ   = 2048;
constexpr int HID   = 1024;
constexpr int NLAY  = 4;
constexpr int MTOT  = BATCH * SEQ;          // 8192
constexpr float LN_EPS = 1e-5f;
constexpr float NEGV   = -1e9f;

// ---------------- WMMA types ----------------
typedef __attribute__((ext_vector_type(16))) __bf16 v16bf;
typedef __attribute__((ext_vector_type(8)))  float  v8f;

__device__ inline v8f wmma_bf16(v16bf a, v16bf b, v8f c) {
  return __builtin_amdgcn_wmma_f32_16x16x32_bf16(
      /*neg_a=*/false, a, /*neg_b=*/false, b,
      /*c_mod=*/(short)0, c, /*reuse_a=*/false, /*reuse_b=*/false);
}

// ---------------- GEMM tiling ----------------
constexpr int BDIM = 256;     // 8 waves (wave32): 4 wave-rows x 2 wave-cols
constexpr int BM = 128, BN = 128, BK = 32;
constexpr int LDT = 40;       // 80 B row stride -> every 16 B chunk aligned

// async 16-byte global -> LDS copy (GLOBAL_LOAD_ASYNC_TO_LDS_B128, ASYNCcnt)
__device__ inline void async_copy16(const void* gsrc, void* lds_dst) {
  const uint32_t l = (uint32_t)(uintptr_t)lds_dst;   // low 32 bits = LDS offset
  asm volatile("global_load_async_to_lds_b128 %0, %1, off"
               :: "v"(l), "v"(gsrc) : "memory");
}
__device__ inline void wait_async() {
  asm volatile("s_wait_asynccnt 0x0" ::: "memory");
}

// Fill one [128 x BK] bf16 tile (g pre-offset to tile row 0; ldg = row stride).
__device__ inline void fill_tile_async(const __bf16* g, size_t ldg, int kb,
                                       __bf16* s, int tid) {
#pragma unroll
  for (int t = 0; t < 2; ++t) {
    const int idx = t * BDIM + tid;      // 0..511 chunks of 8 bf16
    const int row = idx >> 2;            // 4 chunks per 32-elem row
    const int sub = idx & 3;
    async_copy16(g + (size_t)row * ldg + kb + sub * 8,
                 s + row * LDT + sub * 8);
  }
}

// A fragment: 16x32 bf16 (ISA 7.12.2): lane&15 = M row, lane>>4 = K-half,
// dword j<4 -> K base 0, j>=4 -> K base 16; each dword = 2 consecutive K.
__device__ inline v16bf frag_a(const __bf16* As, int mBase, int lane) {
  union { v16bf v; uint32_t u[8]; } r;
  const uint32_t* p = (const uint32_t*)As;
  const int rb   = (mBase + (lane & 15)) * (LDT / 2);
  const int half = lane >> 4;
#pragma unroll
  for (int j = 0; j < 8; ++j) {
    const int kk = ((j & 4) << 2) + half * 8 + (j & 3) * 2;
    r.u[j] = p[rb + (kk >> 1)];
  }
  return r.v;
}

// B fragment: 32x16 bf16: lane&15 = N col; same K order as A (dword j<4 -> K base 0, j>=4 -> 16, +8*(lane>>4)).
__device__ inline v16bf frag_b(const __bf16* Bs, int nBase, int lane) {
  union { v16bf v; uint32_t u[8]; } r;
  const uint32_t* p = (const uint32_t*)Bs;
  const int cb = (nBase + (lane & 15)) * (LDT / 2) + (lane >> 4) * 4;
#pragma unroll
  for (int j = 0; j < 8; ++j) r.u[j] = p[cb + ((j & 4) << 1) + (j & 3)];
  return r.v;
}

// Unified NT GEMM: A[M][K] bf16 (K contiguous), B[N][K] bf16 (K contiguous).
// MODE 0: +bias, out bf16.  MODE 1: *scale + mask -> NEG, out f32.
// MODE 2: in-place residual: Cf32[m][n] += acc.
// Double-buffered LDS; async copies overlapped with WMMA.
template <int MODE>
__global__ __launch_bounds__(BDIM) void gemm_bf16_wmma(
    const __bf16* __restrict__ Ag, const __bf16* __restrict__ Bg,
    const float* __restrict__ bias, const int* __restrict__ mask,
    void* Cg, int M, int N, int K, float scale)
{
  __shared__ __align__(16) __bf16 As[2][BM * LDT];
  __shared__ __align__(16) __bf16 Bs[2][BN * LDT];

  const int tid  = threadIdx.x;
  const int lane = tid & 31;
  const int wave = tid >> 5;
  const int wm   = (wave >> 1) * 32;     // 0,32,64,96
  const int wn   = (wave & 1) * 64;      // 0,64
  const int mb   = blockIdx.y * BM;
  const int nb   = blockIdx.x * BN;
  const __bf16* Abase = Ag + (size_t)mb * K;
  const __bf16* Bbase = Bg + (size_t)nb * K;

  const v8f vzero = {0.f, 0.f, 0.f, 0.f, 0.f, 0.f, 0.f, 0.f};
  v8f acc[2][4];
#pragma unroll
  for (int i = 0; i < 2; ++i)
#pragma unroll
    for (int j = 0; j < 4; ++j) acc[i][j] = vzero;

  const int nk = K / BK;
  // prologue: stage tile 0
  fill_tile_async(Abase, K, 0, As[0], tid);
  fill_tile_async(Bbase, K, 0, Bs[0], tid);
  wait_async();
  __syncthreads();

  for (int it = 0; it < nk; ++it) {
    const int cur = it & 1, nxt = cur ^ 1;
    // stage next tile while computing on current
    if (it + 1 < nk) {
      const int kb = (it + 1) * BK;
      fill_tile_async(Abase, K, kb, As[nxt], tid);
      fill_tile_async(Bbase, K, kb, Bs[nxt], tid);
    }

    const v16bf a0 = frag_a(As[cur], wm,      lane);
    const v16bf a1 = frag_a(As[cur], wm + 16, lane);
    const v16bf b0 = frag_b(Bs[cur], wn,      lane);
    const v16bf b1 = frag_b(Bs[cur], wn + 16, lane);
    const v16bf b2 = frag_b(Bs[cur], wn + 32, lane);
    const v16bf b3 = frag_b(Bs[cur], wn + 48, lane);
    acc[0][0] = wmma_bf16(a0, b0, acc[0][0]);
    acc[0][1] = wmma_bf16(a0, b1, acc[0][1]);
    acc[0][2] = wmma_bf16(a0, b2, acc[0][2]);
    acc[0][3] = wmma_bf16(a0, b3, acc[0][3]);
    acc[1][0] = wmma_bf16(a1, b0, acc[1][0]);
    acc[1][1] = wmma_bf16(a1, b1, acc[1][1]);
    acc[1][2] = wmma_bf16(a1, b2, acc[1][2]);
    acc[1][3] = wmma_bf16(a1, b3, acc[1][3]);

    wait_async();        // next tile landed (this wave)
    __syncthreads();     // all waves done writing nxt / reading cur
  }

  // C layout: dword p -> row (lane>>4)*8 + p, col = lane&15
  const int half = lane >> 4, nl = lane & 15;
#pragma unroll
  for (int i = 0; i < 2; ++i) {
#pragma unroll
    for (int j = 0; j < 4; ++j) {
      const int n = nb + wn + j * 16 + nl;
#pragma unroll
      for (int p = 0; p < 8; ++p) {
        const int m = mb + wm + i * 16 + half * 8 + p;
        float c = acc[i][j][p];
        if constexpr (MODE == 0) {
          c += bias[n];
          ((__bf16*)Cg)[(size_t)m * N + n] = (__bf16)c;
        } else if constexpr (MODE == 1) {
          c *= scale;
          if (mask[n] == 0) c = NEGV;
          ((float*)Cg)[(size_t)m * N + n] = c;
        } else {
          float* cp = (float*)Cg + (size_t)m * N + n;
          *cp = c + *cp;                   // in-place residual add
        }
      }
    }
  }
}

// ---------------- row softmax: f32 scores -> bf16 probs ----------------
__global__ __launch_bounds__(256) void softmax_kernel(float* __restrict__ sc,
                                                      __bf16* __restrict__ pb,
                                                      int n) {
  const int row = blockIdx.x;
  float* pr = sc + (size_t)row * n;
  __bf16* po = pb + (size_t)row * n;
  __shared__ float red[256];
  float mx = -3.0e38f;
  for (int i = threadIdx.x; i < n; i += 256) mx = fmaxf(mx, pr[i]);
  red[threadIdx.x] = mx;
  __syncthreads();
  for (int s = 128; s > 0; s >>= 1) {
    if (threadIdx.x < s) red[threadIdx.x] = fmaxf(red[threadIdx.x], red[threadIdx.x + s]);
    __syncthreads();
  }
  const float m = red[0];
  __syncthreads();
  float sum = 0.f;
  for (int i = threadIdx.x; i < n; i += 256) {
    const float e = __expf(pr[i] - m);
    pr[i] = e;
    sum += e;
  }
  red[threadIdx.x] = sum;
  __syncthreads();
  for (int s = 128; s > 0; s >>= 1) {
    if (threadIdx.x < s) red[threadIdx.x] += red[threadIdx.x + s];
    __syncthreads();
  }
  const float inv = 1.f / red[0];
  for (int i = threadIdx.x; i < n; i += 256) po[i] = (__bf16)(pr[i] * inv);
}

// ---------------- in-place LayerNorm; also emits bf16 x ----------------
__global__ __launch_bounds__(256) void ln_kernel(const float* __restrict__ yin,
                                                 const float* __restrict__ g,
                                                 const float* __restrict__ bta,
                                                 float* out, __bf16* xb) {
  const int row = blockIdx.x;
  const float* py = yin + (size_t)row * HID;
  float v[4], s1 = 0.f, s2 = 0.f;
#pragma unroll
  for (int t = 0; t < 4; ++t) {
    const float x = py[threadIdx.x + t * 256];
    v[t] = x; s1 += x; s2 += x * x;
  }
  __shared__ float r1[256], r2[256];
  r1[threadIdx.x] = s1; r2[threadIdx.x] = s2;
  __syncthreads();
  for (int s = 128; s > 0; s >>= 1) {
    if (threadIdx.x < s) {
      r1[threadIdx.x] += r1[threadIdx.x + s];
      r2[threadIdx.x] += r2[threadIdx.x + s];
    }
    __syncthreads();
  }
  const float mu  = r1[0] * (1.f / HID);
  const float var = r2[0] * (1.f / HID) - mu * mu;
  const float inv = rsqrtf(var + LN_EPS);
#pragma unroll
  for (int t = 0; t < 4; ++t) {
    const int h = threadIdx.x + t * 256;
    const float o = g[h] * (v[t] - mu) * inv + bta[h];
    out[(size_t)row * HID + h] = o;
    xb[(size_t)row * HID + h] = (__bf16)o;
  }
}

// ---------------- conversions ----------------
__global__ void cvt_f32_bf16(const float* __restrict__ s, __bf16* __restrict__ d, int n) {
  const int i = blockIdx.x * blockDim.x + threadIdx.x;
  if (i < n) d[i] = (__bf16)s[i];
}
__global__ void copy_x(const float* __restrict__ s, float* __restrict__ df,
                       __bf16* __restrict__ db, int n) {
  const int i = blockIdx.x * blockDim.x + threadIdx.x;
  if (i < n) { const float v = s[i]; df[i] = v; db[i] = (__bf16)v; }
}

// ---------------- V transpose: [S][H] -> [H][S] bf16, per batch ----------------
__global__ __launch_bounds__(256) void transpose_v(const __bf16* __restrict__ in,
                                                   __bf16* __restrict__ out) {
  __shared__ __bf16 t[32][33];
  const int bx = blockIdx.x * 32;                      // h
  const int by = blockIdx.y * 32;                      // s
  const size_t bo = (size_t)blockIdx.z * SEQ * HID;
#pragma unroll
  for (int r = 0; r < 32; r += 8)
    t[threadIdx.y + r][threadIdx.x] =
        in[bo + (size_t)(by + threadIdx.y + r) * HID + bx + threadIdx.x];
  __syncthreads();
#pragma unroll
  for (int r = 0; r < 32; r += 8)
    out[bo + (size_t)(bx + threadIdx.y + r) * SEQ + by + threadIdx.x] =
        t[threadIdx.x][threadIdx.y + r];
}

// ---------------- host orchestration ----------------
extern "C" void kernel_launch(void* const* d_in, const int* in_sizes, int n_in,
                              void* d_out, int out_size, void* d_ws, size_t ws_size,
                              hipStream_t stream) {
  (void)in_sizes; (void)n_in; (void)out_size; (void)ws_size;
  const float* x    = (const float*)d_in[0];
  const int*   mask = (const int*)d_in[1];
  const float* Wq   = (const float*)d_in[2];
  const float* bq   = (const float*)d_in[3];
  const float* Wk   = (const float*)d_in[4];
  const float* bk   = (const float*)d_in[5];
  const float* Wv   = (const float*)d_in[6];
  const float* bv   = (const float*)d_in[7];
  const float* lnw  = (const float*)d_in[8];
  const float* lnb  = (const float*)d_in[9];

  char* ws = (char*)d_ws;
  size_t off = 0;
  auto alloc = [&](size_t bytes) {
    void* p = ws + off;
    off = (off + bytes + 255) & ~(size_t)255;
    return p;
  };
  float*  xf  = (float*) alloc((size_t)MTOT * HID * 4);     // x (f32), y in place
  __bf16* xb  = (__bf16*)alloc((size_t)MTOT * HID * 2);     // x (bf16)
  __bf16* qb  = (__bf16*)alloc((size_t)MTOT * HID * 2);
  __bf16* kb2 = (__bf16*)alloc((size_t)MTOT * HID * 2);
  __bf16* vb  = (__bf16*)alloc((size_t)MTOT * HID * 2);
  __bf16* vt  = (__bf16*)alloc((size_t)MTOT * HID * 2);     // V^T per batch
  __bf16* wqb = (__bf16*)alloc((size_t)NLAY * HID * HID * 2);
  __bf16* wkb = (__bf16*)alloc((size_t)NLAY * HID * HID * 2);
  __bf16* wvb = (__bf16*)alloc((size_t)NLAY * HID * HID * 2);
  float*  sc  = (float*) alloc((size_t)SEQ * SEQ * 4);      // per-batch scores
  __bf16* pb  = (__bf16*)alloc((size_t)SEQ * SEQ * 2);      // per-batch probs

  const int ntot = MTOT * HID;
  const int nw   = NLAY * HID * HID;
  copy_x<<<(ntot + 255) / 256, 256, 0, stream>>>(x, xf, xb, ntot);
  cvt_f32_bf16<<<(nw + 255) / 256, 256, 0, stream>>>(Wq, wqb, nw);
  cvt_f32_bf16<<<(nw + 255) / 256, 256, 0, stream>>>(Wk, wkb, nw);
  cvt_f32_bf16<<<(nw + 255) / 256, 256, 0, stream>>>(Wv, wvb, nw);

  const dim3 blk(BDIM);
  const dim3 gQKV(HID / BN, MTOT / BM);   // 8 x 64
  const dim3 gSC (SEQ / BN, SEQ / BM);    // 16 x 16
  const dim3 gPV (HID / BN, SEQ / BM);    // 8 x 16
  const dim3 gTR (HID / 32, SEQ / 32, BATCH);
  const dim3 bTR (32, 8);
  const float inv_sqrt_h = 0.03125f;      // 1/sqrt(1024)

  for (int l = 0; l < NLAY; ++l) {
    const size_t wo = (size_t)l * HID * HID;
    const size_t bo = (size_t)l * HID;
    gemm_bf16_wmma<0><<<gQKV, blk, 0, stream>>>(
        xb, wqb + wo, bq + bo, nullptr, qb, MTOT, HID, HID, 1.f);
    gemm_bf16_wmma<0><<<gQKV, blk, 0, stream>>>(
        xb, wkb + wo, bk + bo, nullptr, kb2, MTOT, HID, HID, 1.f);
    gemm_bf16_wmma<0><<<gQKV, blk, 0, stream>>>(
        xb, wvb + wo, bv + bo, nullptr, vb, MTOT, HID, HID, 1.f);
    transpose_v<<<gTR, bTR, 0, stream>>>(vb, vt);

    for (int b = 0; b < BATCH; ++b) {
      const size_t ao = (size_t)b * SEQ * HID;
      gemm_bf16_wmma<1><<<gSC, blk, 0, stream>>>(
          qb + ao, kb2 + ao, nullptr, mask + (size_t)b * SEQ, sc,
          SEQ, SEQ, HID, inv_sqrt_h);
      softmax_kernel<<<SEQ, 256, 0, stream>>>(sc, pb, SEQ);
      gemm_bf16_wmma<2><<<gPV, blk, 0, stream>>>(
          pb, vt + ao, nullptr, nullptr, xf + ao, SEQ, HID, SEQ, 1.f);
    }

    float* outx = (l == NLAY - 1) ? (float*)d_out : xf;
    ln_kernel<<<MTOT, 256, 0, stream>>>(xf, lnw + bo, lnb + bo, outx, xb);
  }
}
